// LSTMAutoencoder_12618613916355
// MI455X (gfx1250) — matrix operand order repaired and checked by reading, not compiled, not run on hardware
//
#include <hip/hip_runtime.h>
#include <hip/hip_bf16.h>
#include <math.h>

typedef __hip_bfloat16 bf16_t;
typedef __attribute__((ext_vector_type(16))) __bf16 v16bf;
typedef __attribute__((ext_vector_type(8)))  float  v8f;

#define BATCH 256
#define TSTEPS 256
#define INDIM 64
#define HDIM 512
#define KENC 576   // INDIM + HDIM
#define KDEC 512   // (dec_Wx + dec_Wh) folded -> K = HDIM

union FragU { v16bf v; uint4 q[2]; };

// A-matrix (weights) fragment, 16x32, contiguous k order: lane half h holds k0+16h .. k0+16h+15.
// rowp must already include the lane-dependent +half*16 term, so each load
// here is base + compile-time constant -> immediate-offset global_load_b128.
__device__ inline v16bf load_w_frag(const bf16_t* rowp, int k0) {
    FragU u;
    u.q[0] = *(const uint4*)(rowp + k0);
    u.q[1] = *(const uint4*)(rowp + k0 + 8);
    return u.v;
}

// B-matrix (activations) fragment, 32x16, same contiguous k order as the A fragment.
// rowp must already include the lane-dependent +half*16 term.
__device__ inline v16bf load_act_frag(const bf16_t* rowp, int k0) {
    FragU u;
    u.q[0] = *(const uint4*)(rowp + k0);
    u.q[1] = *(const uint4*)(rowp + k0 + 8);
    return u.v;
}

__device__ inline v8f wmma_bf16(v16bf a, v16bf b, v8f c) {
    return __builtin_amdgcn_wmma_f32_16x16x32_bf16(
        false, a, false, b, (short)0, c, false, false);
}

__device__ inline float sigmoidf_fast(float x) {
    return 1.0f / (1.0f + __expf(-x));
}

// Load one K-slice worth of fragments (2 A-rows, 2 B-rows) into caller regs.
template <int KTOT, bool ENC>
__device__ inline void load_gate_frags(const bf16_t* __restrict__ arow0,
                                       const bf16_t* __restrict__ arow1,
                                       const bf16_t* __restrict__ xrow0,
                                       const bf16_t* __restrict__ xrow1,
                                       const bf16_t* __restrict__ hrow0,
                                       const bf16_t* __restrict__ hrow1,
                                       int k0,
                                       v16bf& a0, v16bf& a1,
                                       v16bf& b0, v16bf& b1)
{
    a0 = load_w_frag(arow0, k0);
    a1 = load_w_frag(arow1, k0);
    if (ENC && k0 < INDIM) {
        b0 = load_act_frag(xrow0, k0);
        b1 = load_act_frag(xrow1, k0);
    } else {
        const int kh = ENC ? (k0 - INDIM) : k0;
        b0 = load_act_frag(hrow0, kh);
        b1 = load_act_frag(hrow1, kh);
    }
}

// ---------------------------------------------------------------------------
// One LSTM step: gates = A * concat(x_t, h)  (or A*h for decoder), then
// elementwise state update. Grid: 64 WGs (16 h-tiles x 4 b-tiles), 256 thr.
// Each WG: 32(h) x 64(b) tile, all 4 gates. Wave w: gate w&3, b-half w>>2,
// computing a 32x32 subtile as 2x2 WMMA accumulators. The K-loop is
// double-buffered so the next slice's 8 b128 loads overlap current WMMAs.
// ---------------------------------------------------------------------------
template <int KTOT, bool ENC>
__device__ void lstm_gate_step(const bf16_t* __restrict__ A,
                               const float*  __restrict__ bias,
                               const bf16_t* __restrict__ xb, int t,
                               const bf16_t* __restrict__ h_in,
                               bf16_t* __restrict__ h_out,
                               float*  __restrict__ c)
{
    __shared__ float lds[4][32][64];
    const int tid  = threadIdx.x;
    const int lane = tid & 31;
    const int w    = tid >> 5;
    const int g    = w & 3;       // gate: 0=f 1=i 2=o 3=c~
    const int bh   = w >> 2;      // b half (0/1) of the 64-wide b tile
    const int hTile = blockIdx.x & 15;
    const int bTile = blockIdx.x >> 4;
    const int h0    = hTile * 32;
    const int bBase = bTile * 64;
    const int half  = lane >> 4;
    const int l16   = lane & 15;

    // Accumulators pre-loaded with the folded bias (depends on m only).
    v8f acc[2][2];
#pragma unroll
    for (int mt = 0; mt < 2; ++mt) {
#pragma unroll
        for (int j = 0; j < 8; ++j) {
            float bv = bias[g * HDIM + h0 + mt * 16 + 8 * half + j];
            acc[mt][0][j] = bv;
            acc[mt][1][j] = bv;
        }
    }

    // Base pointers with all lane-dependent terms folded in; every load in
    // the unrolled K-loop below is then base + immediate.
    const bf16_t* arow0 = A + (size_t)(g * HDIM + h0 +      l16) * KTOT + half * 16;
    const bf16_t* arow1 = A + (size_t)(g * HDIM + h0 + 16 + l16) * KTOT + half * 16;
    const int brow0 = bBase + bh * 32 + l16;
    const int brow1 = brow0 + 16;
    const bf16_t* xrow0 = ENC ? xb + ((size_t)brow0 * TSTEPS + t) * INDIM + half * 16 : nullptr;
    const bf16_t* xrow1 = ENC ? xb + ((size_t)brow1 * TSTEPS + t) * INDIM + half * 16 : nullptr;
    const bf16_t* hrow0 = h_in + (size_t)brow0 * HDIM + half * 16;
    const bf16_t* hrow1 = h_in + (size_t)brow1 * HDIM + half * 16;

    // Software-pipelined K loop: frag[cur] consumed while frag[nxt] loads.
    v16bf a0[2], a1[2], b0[2], b1[2];
    load_gate_frags<KTOT, ENC>(arow0, arow1, xrow0, xrow1, hrow0, hrow1,
                               0, a0[0], a1[0], b0[0], b1[0]);
    int cur = 0;
#pragma unroll
    for (int k0 = 0; k0 < KTOT; k0 += 32) {
        const int nxt = cur ^ 1;
        if (k0 + 32 < KTOT)
            load_gate_frags<KTOT, ENC>(arow0, arow1, xrow0, xrow1, hrow0, hrow1,
                                       k0 + 32,
                                       a0[nxt], a1[nxt], b0[nxt], b1[nxt]);
        acc[0][0] = wmma_bf16(a0[cur], b0[cur], acc[0][0]);
        acc[0][1] = wmma_bf16(a0[cur], b1[cur], acc[0][1]);
        acc[1][0] = wmma_bf16(a1[cur], b0[cur], acc[1][0]);
        acc[1][1] = wmma_bf16(a1[cur], b1[cur], acc[1][1]);
        cur = nxt;
    }

    // Exchange gate tiles through LDS (each wave owns one gate).
    // C/D layout: lane l holds n=l%16, m = j + 8*(l/16) in VGPR j.
#pragma unroll
    for (int mt = 0; mt < 2; ++mt)
#pragma unroll
        for (int nt = 0; nt < 2; ++nt)
#pragma unroll
            for (int j = 0; j < 8; ++j) {
                const int hl = mt * 16 + 8 * half + j;
                const int bl = bh * 32 + nt * 16 + l16;
                lds[g][hl][bl] = acc[mt][nt][j];
            }
    __syncthreads();

    // Elementwise state update: 2048 elements over 256 threads.
#pragma unroll
    for (int e = tid; e < 32 * 64; e += 256) {
        const int hl = e >> 6, bl = e & 63;
        const int hg = h0 + hl;
        const int bg = bBase + bl;
        const float fg = sigmoidf_fast(lds[0][hl][bl]);
        const float ig = sigmoidf_fast(lds[1][hl][bl]);
        const float og = sigmoidf_fast(lds[2][hl][bl]);
        const float ct = tanhf(lds[3][hl][bl]);
        const size_t idx = (size_t)bg * HDIM + hg;
        const float cn = fg * c[idx] + ig * ct;
        c[idx] = cn;
        h_out[idx] = __float2bfloat16(og * tanhf(cn));
    }
}

// ---------------------------------------------------------------------------
// Reconstruction: out[b][t][:] = h[b][:] * rec_W^T + rec_b.
// D = recW(64 x 512) x h^T(512 x 256). 4 WGs x 256 thr; wave w: o-tile w&3,
// b-half w>>2, 1x2 WMMA accumulators over a 64(b) slice. Double-buffered.
// ---------------------------------------------------------------------------
__device__ void rec_step(const bf16_t* __restrict__ recWb,
                         const float*  __restrict__ rec_b,
                         const bf16_t* __restrict__ h,
                         float* __restrict__ out, int t, int rb)
{
    const int tid   = threadIdx.x;
    const int lane  = tid & 31;
    const int w     = tid >> 5;
    const int mtile = w & 3;
    const int bhalf = w >> 2;
    const int half  = lane >> 4;
    const int l16   = lane & 15;
    const int b0    = rb * 64 + bhalf * 32;

    v8f acc[2];
#pragma unroll
    for (int j = 0; j < 8; ++j) {
        float bv = rec_b[mtile * 16 + 8 * half + j];
        acc[0][j] = bv;
        acc[1][j] = bv;
    }
    const bf16_t* arow  = recWb + (size_t)(mtile * 16 + l16) * HDIM + half * 16;
    const bf16_t* brow0 = h + (size_t)(b0 + l16) * HDIM + half * 16;
    const bf16_t* brow1 = h + (size_t)(b0 + 16 + l16) * HDIM + half * 16;

    v16bf a[2], bb0[2], bb1[2];
    a[0]   = load_w_frag(arow, 0);
    bb0[0] = load_act_frag(brow0, 0);
    bb1[0] = load_act_frag(brow1, 0);
    int cur = 0;
#pragma unroll
    for (int k0 = 0; k0 < HDIM; k0 += 32) {
        const int nxt = cur ^ 1;
        if (k0 + 32 < HDIM) {
            a[nxt]   = load_w_frag(arow, k0 + 32);
            bb0[nxt] = load_act_frag(brow0, k0 + 32);
            bb1[nxt] = load_act_frag(brow1, k0 + 32);
        }
        acc[0] = wmma_bf16(a[cur], bb0[cur], acc[0]);
        acc[1] = wmma_bf16(a[cur], bb1[cur], acc[1]);
        cur = nxt;
    }
#pragma unroll
    for (int nt = 0; nt < 2; ++nt)
#pragma unroll
        for (int j = 0; j < 8; ++j) {
            const int o = mtile * 16 + 8 * half + j;
            const int b = b0 + nt * 16 + l16;
            out[((size_t)b * TSTEPS + t) * INDIM + o] = acc[nt][j];
        }
}

// ---------------------------------------------------------------------------
// Kernels
// ---------------------------------------------------------------------------
__global__ __launch_bounds__(256)
void enc_step_kernel(const bf16_t* A, const float* bias, const bf16_t* xb,
                     const bf16_t* h_in, bf16_t* h_out, float* c, int t)
{
    lstm_gate_step<KENC, true>(A, bias, xb, t, h_in, h_out, c);
}

__global__ __launch_bounds__(256)
void dec_step_kernel(const bf16_t* A, const float* bias,
                     const bf16_t* recWb, const float* rec_b,
                     const bf16_t* h_in, bf16_t* h_out, float* c,
                     float* out, int k)
{
    if (blockIdx.x < 64) {
        lstm_gate_step<KDEC, false>(A, bias, nullptr, 0, h_in, h_out, c);
    } else {
        if (k == 0) return;                  // h_0 = h_enc is not part of hs
        rec_step(recWb, rec_b, h_in, out, k - 1, blockIdx.x - 64);
    }
}

__global__ __launch_bounds__(256)
void rec_final_kernel(const bf16_t* recWb, const float* rec_b,
                      const bf16_t* h, float* out)
{
    rec_step(recWb, rec_b, h, out, TSTEPS - 1, blockIdx.x);
}

// Pack weights (bf16, concat over K), fold biases, convert x, zero state.
__global__ __launch_bounds__(256)
void prep_kernel(const float* x,
                 const float* enc_Wx, const float* enc_bWx,
                 const float* enc_Wh, const float* enc_bWh, const float* enc_b,
                 const float* dec_Wx, const float* dec_bWx,
                 const float* dec_Wh, const float* dec_bWh, const float* dec_b,
                 const float* rec_W,
                 bf16_t* A_enc, bf16_t* A_dec, bf16_t* recWb, bf16_t* xb,
                 float* bias_enc, float* bias_dec,
                 bf16_t* h0, bf16_t* h1, float* c)
{
    const int stride = gridDim.x * blockDim.x;
    const int tid = blockIdx.x * blockDim.x + threadIdx.x;

    for (int i = tid; i < 4 * HDIM * KENC; i += stride) {
        const int r = i / KENC, k = i % KENC;
        const float v = (k < INDIM) ? enc_Wx[r * INDIM + k]
                                    : enc_Wh[r * HDIM + (k - INDIM)];
        A_enc[i] = __float2bfloat16(v);
    }
    for (int i = tid; i < 4 * HDIM * KDEC; i += stride)
        A_dec[i] = __float2bfloat16(dec_Wx[i] + dec_Wh[i]);   // x input == h
    for (int i = tid; i < INDIM * HDIM; i += stride)
        recWb[i] = __float2bfloat16(rec_W[i]);
    for (int i = tid; i < BATCH * TSTEPS * INDIM; i += stride)
        xb[i] = __float2bfloat16(x[i]);
    for (int i = tid; i < 4 * HDIM; i += stride) {
        bias_enc[i] = enc_bWx[i] + enc_bWh[i] + enc_b[i];
        bias_dec[i] = dec_bWx[i] + dec_bWh[i] + dec_b[i];
    }
    for (int i = tid; i < BATCH * HDIM; i += stride) {
        h0[i] = __float2bfloat16(0.0f);
        h1[i] = __float2bfloat16(0.0f);
        c[i]  = 0.0f;
    }
}

// ---------------------------------------------------------------------------
// Host entry
// ---------------------------------------------------------------------------
extern "C" void kernel_launch(void* const* d_in, const int* in_sizes, int n_in,
                              void* d_out, int out_size, void* d_ws, size_t ws_size,
                              hipStream_t stream)
{
    (void)in_sizes; (void)n_in; (void)out_size; (void)ws_size;
    const float* x       = (const float*)d_in[0];
    const float* enc_Wx  = (const float*)d_in[1];
    const float* enc_bWx = (const float*)d_in[2];
    const float* enc_Wh  = (const float*)d_in[3];
    const float* enc_bWh = (const float*)d_in[4];
    const float* enc_b   = (const float*)d_in[5];
    const float* dec_Wx  = (const float*)d_in[6];
    const float* dec_bWx = (const float*)d_in[7];
    const float* dec_Wh  = (const float*)d_in[8];
    const float* dec_bWh = (const float*)d_in[9];
    const float* dec_b   = (const float*)d_in[10];
    const float* rec_W   = (const float*)d_in[11];
    const float* rec_b   = (const float*)d_in[12];
    float* out = (float*)d_out;

    char* ws = (char*)d_ws;
    size_t off = 0;
    auto take = [&](size_t bytes) -> void* {
        void* p = (void*)(ws + off);
        off += (bytes + 255) & ~(size_t)255;
        return p;
    };
    bf16_t* A_enc    = (bf16_t*)take((size_t)4 * HDIM * KENC * 2);
    bf16_t* A_dec    = (bf16_t*)take((size_t)4 * HDIM * KDEC * 2);
    bf16_t* recWb    = (bf16_t*)take((size_t)INDIM * HDIM * 2);
    bf16_t* xb       = (bf16_t*)take((size_t)BATCH * TSTEPS * INDIM * 2);
    float*  bias_enc = (float*) take((size_t)4 * HDIM * 4);
    float*  bias_dec = (float*) take((size_t)4 * HDIM * 4);
    bf16_t* hbuf[2];
    hbuf[0]          = (bf16_t*)take((size_t)BATCH * HDIM * 2);
    hbuf[1]          = (bf16_t*)take((size_t)BATCH * HDIM * 2);
    float*  cbuf     = (float*) take((size_t)BATCH * HDIM * 4);

    prep_kernel<<<512, 256, 0, stream>>>(
        x, enc_Wx, enc_bWx, enc_Wh, enc_bWh, enc_b,
        dec_Wx, dec_bWx, dec_Wh, dec_bWh, dec_b, rec_W,
        A_enc, A_dec, recWb, xb, bias_enc, bias_dec,
        hbuf[0], hbuf[1], cbuf);

    // Encoder scan: h_0 = 0 in hbuf[0]; step t reads hbuf[t&1], writes other.
    for (int t = 0; t < TSTEPS; ++t)
        enc_step_kernel<<<64, 256, 0, stream>>>(
            A_enc, bias_enc, xb, hbuf[t & 1], hbuf[(t + 1) & 1], cbuf, t);

    // Decoder scan (h_enc lands in hbuf[0]); rec for step k-1 fused into
    // launch k (it only needs h_k = this launch's input).
    for (int k = 0; k < TSTEPS; ++k)
        dec_step_kernel<<<68, 256, 0, stream>>>(
            A_dec, bias_dec, recWb, rec_b,
            hbuf[k & 1], hbuf[(k + 1) & 1], cbuf, out, k);

    // Final reconstruction for the last decoder hidden state (in hbuf[0]).
    rec_final_kernel<<<4, 256, 0, stream>>>(recWb, rec_b, hbuf[0], out);
}
